// PAM_Module_v3_42872363549059
// MI455X (gfx1250) — compile-verified, matrix operand order checked by reading, not run on hardware
//
#include <hip/hip_runtime.h>
#include <stdint.h>

typedef __attribute__((ext_vector_type(16))) __bf16 v16bf;
typedef __attribute__((ext_vector_type(8)))  float  v8f;

#define B_    4
#define C_    512
#define D_    64
#define N_    4096
#define BM    128
#define BN    64
#define BK    32
#define NT    4       // BN/16 subtiles per wave
#define KS    16      // K-split factor for V*Q^T
#define KCH   (N_ / KS)

// f32 -> bf16 round-to-nearest-even (bit pattern)
__device__ __forceinline__ unsigned short f2bf(float f) {
  unsigned int u = __float_as_uint(f);
  u += 0x7FFFu + ((u >> 16) & 1u);
  return (unsigned short)(u >> 16);
}

union Frag { v16bf v; unsigned int u[8]; };

// Build a 16x32 bf16 operand fragment from an LDS row of 32 bf16 (k-major).
// lanes 0-15: K = {0..7,16..23}; lanes 16-31: K = {8..15,24..31}
__device__ __forceinline__ v16bf load_frag(const unsigned short* row32, int half) {
  Frag f;
  const unsigned int* p = (const unsigned int*)row32;
#pragma unroll
  for (int j = 0; j < 8; ++j)
    f.u[j] = p[(j & 3) + ((j >> 2) << 3) + (half << 2)];
  return f.v;
}

#define WMMA_BF16(A, Bf, Cacc) \
  __builtin_amdgcn_wmma_f32_16x16x32_bf16(false, (A), false, (Bf), (short)0, (Cacc), false, false)

// ---------------------------------------------------------------------------
// Kernel 1: fused QKV projection.  Wall[640x512]=[wq;wk;wv];  out = Wall*x + b
// ---------------------------------------------------------------------------
__global__ __launch_bounds__(256) void pam_qkv_proj(
    const float* __restrict__ x,
    const float* __restrict__ wq, const float* __restrict__ bq,
    const float* __restrict__ wk, const float* __restrict__ bk,
    const float* __restrict__ wvp, const float* __restrict__ bv,
    unsigned short* __restrict__ qws, unsigned short* __restrict__ kws,
    unsigned short* __restrict__ vws) {
  __shared__ unsigned short As[BM][BK];   // [row][k]
  __shared__ unsigned short Bs[BN][BK];   // [pixel][k]

  const int t = threadIdx.x;
  const int lane = t & 31, wid = t >> 5;
  const int half = lane >> 4, l15 = lane & 15;
  const int m0 = blockIdx.x * BM;
  const int n0 = blockIdx.y * BN;
  const int b  = blockIdx.z;
  const float* xb = x + (size_t)b * C_ * N_;

  v8f acc[NT] = {};

  const int a_row = t >> 1, a_seg = (t & 1) * 16;  // A: 2 thr/row, 16 f32 each
  const int b_k   = t >> 3, b_seg = (t & 7) * 8;   // B: 8 thr/k-row, 8 pixels each

  const int o = m0 + a_row;
  const float* wrow;
  if (o < 64)       wrow = wq  + (size_t)o * C_;
  else if (o < 128) wrow = wk  + (size_t)(o - 64) * C_;
  else              wrow = wvp + (size_t)(o - 128) * C_;

  for (int k0 = 0; k0 < C_; k0 += BK) {
#pragma unroll
    for (int j = 0; j < 16; ++j)
      As[a_row][a_seg + j] = f2bf(wrow[k0 + a_seg + j]);
    const float* xr = xb + (size_t)(k0 + b_k) * N_ + n0 + b_seg;
#pragma unroll
    for (int j = 0; j < 8; ++j)
      Bs[b_seg + j][b_k] = f2bf(xr[j]);            // transpose into [pixel][k]
    // pull next K-slab toward the WGP while we compute (locality 3 => WGP scope)
    __builtin_prefetch(wrow + k0 + BK + a_seg, 0, 3);
    __builtin_prefetch(xr + (size_t)BK * N_, 0, 3);
    __syncthreads();

    v16bf a  = load_frag(&As[wid * 16 + l15][0], half);
    v16bf b0 = load_frag(&Bs[ 0 + l15][0], half);
    v16bf b1 = load_frag(&Bs[16 + l15][0], half);
    v16bf b2 = load_frag(&Bs[32 + l15][0], half);
    v16bf b3 = load_frag(&Bs[48 + l15][0], half);
    acc[0] = WMMA_BF16(a, b0, acc[0]);
    acc[1] = WMMA_BF16(a, b1, acc[1]);
    acc[2] = WMMA_BF16(a, b2, acc[2]);
    acc[3] = WMMA_BF16(a, b3, acc[3]);
    __syncthreads();
  }

#pragma unroll
  for (int r = 0; r < 8; ++r) {
    const int orow = m0 + wid * 16 + r + half * 8;
    float bias; unsigned short* dst;
    if (orow < 64)       { bias = bq[orow];       dst = qws + ((size_t)b * D_ + orow)        * N_; }
    else if (orow < 128) { bias = bk[orow - 64];  dst = kws + ((size_t)b * D_ + (orow - 64)) * N_; }
    else                 { bias = bv[orow - 128]; dst = vws + ((size_t)b * C_ + (orow - 128))* N_; }
#pragma unroll
    for (int nt = 0; nt < NT; ++nt)
      dst[n0 + nt * 16 + l15] = f2bf(acc[nt][r] + bias);
  }
}

// ---------------------------------------------------------------------------
// Kernel 2: Mpart[b][ks] = V[C,Kchunk] * Q[d,Kchunk]^T
// bf16 tiles copied with gfx1250 async global->LDS engine, double-buffered.
// 3 async ops/wave per slab; in-order completion => s_wait_asynccnt 3 pins
// the previous slab while the next streams in under the WMMAs.
// ---------------------------------------------------------------------------
__global__ __launch_bounds__(256) void pam_vqt(
    const unsigned short* __restrict__ vws,
    const unsigned short* __restrict__ qws,
    float* __restrict__ Mpart) {
  __shared__ unsigned short As[2][BM][BK];   // [buf][c-row][k]
  __shared__ unsigned short Bs[2][BN][BK];   // [buf][dd-col][k]

  const int t = threadIdx.x;
  const int lane = t & 31, wid = t >> 5;
  const int half = lane >> 4, l15 = lane & 15;
  const int m0 = blockIdx.x * BM;
  const int kbase = blockIdx.y * KCH;
  const int b = blockIdx.z;
  const unsigned short* vb = vws + (size_t)b * C_ * N_;
  const unsigned short* qb = qws + (size_t)b * D_ * N_;
  float* Mp = Mpart + ((size_t)(b * KS + blockIdx.y)) * C_ * D_;

  v8f acc[NT] = {};
  const int a_row = t >> 1, a_seg = (t & 1) * 16;  // 32B per thread (2x b128)
  const int b_row = t >> 2, b_seg = (t & 3) * 8;   // 16B per thread (1x b128)

  const unsigned short* vsrc = vb + (size_t)(m0 + a_row) * N_ + kbase + a_seg;
  const unsigned short* qsrc = qb + (size_t)b_row * N_ + kbase + b_seg;
  const unsigned ldsA[2] = { (unsigned)(uintptr_t)&As[0][a_row][a_seg],
                             (unsigned)(uintptr_t)&As[1][a_row][a_seg] };
  const unsigned ldsB[2] = { (unsigned)(uintptr_t)&Bs[0][b_row][b_seg],
                             (unsigned)(uintptr_t)&Bs[1][b_row][b_seg] };

  auto prefetch = [&](int buf, int kk) {
    const unsigned short* vp = vsrc + kk;
    const unsigned short* qp = qsrc + kk;
    asm volatile("global_load_async_to_lds_b128 %0, %1, off"
                 :: "v"(ldsA[buf]), "v"(vp) : "memory");
    asm volatile("global_load_async_to_lds_b128 %0, %1, off offset:16"
                 :: "v"(ldsA[buf]), "v"(vp) : "memory");   // offset applies to LDS & mem
    asm volatile("global_load_async_to_lds_b128 %0, %1, off"
                 :: "v"(ldsB[buf]), "v"(qp) : "memory");
  };

  prefetch(0, 0);
  int buf = 0;
  for (int kk = 0; kk < KCH; kk += BK) {
    const bool more = (kk + BK) < KCH;
    if (more) {
      prefetch(buf ^ 1, kk + BK);                 // buf^1 free: barrier of prev iter
      asm volatile("s_wait_asynccnt 0x3" ::: "memory");
    } else {
      asm volatile("s_wait_asynccnt 0x0" ::: "memory");
    }
    __syncthreads();

    const unsigned short (*Asb)[BK] = As[buf];
    const unsigned short (*Bsb)[BK] = Bs[buf];
    v16bf a  = load_frag(&Asb[wid * 16 + l15][0], half);
    v16bf b0 = load_frag(&Bsb[ 0 + l15][0], half);
    v16bf b1 = load_frag(&Bsb[16 + l15][0], half);
    v16bf b2 = load_frag(&Bsb[32 + l15][0], half);
    v16bf b3 = load_frag(&Bsb[48 + l15][0], half);
    acc[0] = WMMA_BF16(a, b0, acc[0]);
    acc[1] = WMMA_BF16(a, b1, acc[1]);
    acc[2] = WMMA_BF16(a, b2, acc[2]);
    acc[3] = WMMA_BF16(a, b3, acc[3]);
    __syncthreads();                              // buf read-done before overwrite
    buf ^= 1;
  }

#pragma unroll
  for (int r = 0; r < 8; ++r) {
    const int c = m0 + wid * 16 + r + half * 8;
#pragma unroll
    for (int nt = 0; nt < NT; ++nt)
      Mp[(size_t)c * D_ + nt * 16 + l15] = acc[nt][r];
  }
}

// ---------------------------------------------------------------------------
// Kernel 3: deterministic reduction of K-split partials, scale 1/N -> bf16
// ---------------------------------------------------------------------------
__global__ __launch_bounds__(256) void pam_reduce(
    const float* __restrict__ Mpart, unsigned short* __restrict__ Mbf) {
  const int i = blockIdx.x * blockDim.x + threadIdx.x;   // over B*C*D
  if (i >= B_ * C_ * D_) return;
  const int b = i / (C_ * D_);
  const int cd = i - b * (C_ * D_);
  float s = 0.f;
#pragma unroll
  for (int ks = 0; ks < KS; ++ks)
    s += Mpart[((size_t)(b * KS + ks)) * C_ * D_ + cd];
  Mbf[i] = f2bf(s * (1.0f / (float)N_));
}

// ---------------------------------------------------------------------------
// Kernel 4: out = gamma * (M[C,d] * K[d,N]) + x    (K = 64, 2 WMMA steps)
// ---------------------------------------------------------------------------
__global__ __launch_bounds__(256) void pam_attn_out(
    const unsigned short* __restrict__ Mbf,
    const unsigned short* __restrict__ kws,
    const float* __restrict__ x,
    const float* __restrict__ gamma,
    float* __restrict__ out) {
  __shared__ unsigned short As[BM][BK];   // [c-row][dd]
  __shared__ unsigned short Bs[BN][BK];   // [pixel][dd]

  const int t = threadIdx.x;
  const int lane = t & 31, wid = t >> 5;
  const int half = lane >> 4, l15 = lane & 15;
  const int m0 = blockIdx.x * BM;
  const int n0 = blockIdx.y * BN;
  const int b  = blockIdx.z;
  const unsigned short* Mb = Mbf + (size_t)b * C_ * D_;
  const unsigned short* kb = kws + (size_t)b * D_ * N_;
  const float* xb = x + (size_t)b * C_ * N_;
  float* ob = out + (size_t)b * C_ * N_;

  v8f acc[NT] = {};
  const int a_row = t >> 1, a_seg = (t & 1) * 16;
  const int b_k = t >> 3, b_seg = (t & 7) * 8;

  for (int k0 = 0; k0 < D_; k0 += BK) {
    const unsigned short* ms = Mb + (size_t)(m0 + a_row) * D_ + k0 + a_seg;
#pragma unroll
    for (int j = 0; j < 16; ++j) As[a_row][a_seg + j] = ms[j];
    const unsigned short* ksrc = kb + (size_t)(k0 + b_k) * N_ + n0 + b_seg;
#pragma unroll
    for (int j = 0; j < 8; ++j) Bs[b_seg + j][b_k] = ksrc[j];   // transpose
    if (k0 == 0) {
      __builtin_prefetch(ms + BK, 0, 3);
      __builtin_prefetch(ksrc + (size_t)BK * N_, 0, 3);
    }
    __syncthreads();

    v16bf a  = load_frag(&As[wid * 16 + l15][0], half);
    v16bf b0 = load_frag(&Bs[ 0 + l15][0], half);
    v16bf b1 = load_frag(&Bs[16 + l15][0], half);
    v16bf b2 = load_frag(&Bs[32 + l15][0], half);
    v16bf b3 = load_frag(&Bs[48 + l15][0], half);
    acc[0] = WMMA_BF16(a, b0, acc[0]);
    acc[1] = WMMA_BF16(a, b1, acc[1]);
    acc[2] = WMMA_BF16(a, b2, acc[2]);
    acc[3] = WMMA_BF16(a, b3, acc[3]);
    __syncthreads();
  }

  const float g = gamma[0];
#pragma unroll
  for (int r = 0; r < 8; ++r) {
    const int c = m0 + wid * 16 + r + half * 8;
    const size_t rowoff = (size_t)c * N_ + n0;
#pragma unroll
    for (int nt = 0; nt < NT; ++nt) {
      const int col = nt * 16 + l15;
      ob[rowoff + col] = g * acc[nt][r] + xb[rowoff + col];
    }
  }
}

// ---------------------------------------------------------------------------
extern "C" void kernel_launch(void* const* d_in, const int* in_sizes, int n_in,
                              void* d_out, int out_size, void* d_ws, size_t ws_size,
                              hipStream_t stream) {
  const float* x     = (const float*)d_in[0];
  const float* wq    = (const float*)d_in[1];
  const float* bq    = (const float*)d_in[2];
  const float* wk    = (const float*)d_in[3];
  const float* bk    = (const float*)d_in[4];
  const float* wvp   = (const float*)d_in[5];
  const float* bv    = (const float*)d_in[6];
  const float* gamma = (const float*)d_in[7];
  float* out = (float*)d_out;

  char* ws = (char*)d_ws;
  unsigned short* qws = (unsigned short*)ws;  ws += (size_t)B_ * D_ * N_ * 2;      // 2 MB
  unsigned short* kws = (unsigned short*)ws;  ws += (size_t)B_ * D_ * N_ * 2;      // 2 MB
  unsigned short* vws = (unsigned short*)ws;  ws += (size_t)B_ * C_ * N_ * 2;      // 16 MB
  float*          Mpart = (float*)ws;         ws += (size_t)B_ * KS * C_ * D_ * 4; // 8 MB
  unsigned short* Mbf = (unsigned short*)ws;                                       // 256 KB

  // 1) fused QKV projection (virtual [640 x 512] weight, 5 M-tiles of 128)
  pam_qkv_proj<<<dim3(640 / BM, N_ / BN, B_), 256, 0, stream>>>(
      x, wq, bq, wk, bk, wvp, bv, qws, kws, vws);

  // 2) M partials = V * Q^T over 16 K-chunks (async-to-LDS double buffered)
  pam_vqt<<<dim3(C_ / BM, KS, B_), 256, 0, stream>>>(vws, qws, Mpart);

  // 3) reduce partials, scale 1/N, convert to bf16 (deterministic)
  const int mtot = B_ * C_ * D_;
  pam_reduce<<<(mtot + 255) / 256, 256, 0, stream>>>(Mpart, Mbf);

  // 4) out = gamma * (M * K) + x
  pam_attn_out<<<dim3(C_ / BM, N_ / BN, B_), 256, 0, stream>>>(
      Mbf, kws, x, gamma, out);
}
